// BlockSparseAttentionKernel_66005057405342
// MI455X (gfx1250) — matrix operand order repaired and checked by reading, not compiled, not run on hardware
//
#include <hip/hip_runtime.h>
#include <hip/hip_bf16.h>

// Problem constants (match reference)
#define BB   2
#define SS   2048
#define HH   16
#define DD   64
#define NBQ  32
#define NBK  32
#define KSTR 72   // LDS row stride (half-words) for K/V tiles (pad vs bank conflicts)
#define PSTR 72   // LDS row stride for P staging

typedef __bf16 bf16;
typedef __attribute__((ext_vector_type(16))) __bf16 v16bf;
typedef __attribute__((ext_vector_type(8)))  float  v8f;

union BF16Frag {
    v16bf v;
    bf16  e[16];
    uint4 q[2];
};

union BF16Pack8 {
    uint4 q;
    bf16  e[8];
};

__global__ __launch_bounds__(128)
void bsattn_fa_wmma_kernel(const float* __restrict__ q,
                           const float* __restrict__ k,
                           const float* __restrict__ v,
                           const unsigned char* __restrict__ bmask,
                           float* __restrict__ out) {
    __shared__ bf16 kt[64 * KSTR];       // K tile, bf16, [key][d]
    __shared__ bf16 vt[64 * KSTR];       // V tile, bf16, [d][key] (transposed)
    __shared__ bf16 pst[4 * 16 * PSTR];  // per-wave P staging, bf16, [row][key]

    const int bid = blockIdx.x;
    const int qi  = bid & (NBQ - 1);
    const int h   = (bid >> 5) & (HH - 1);
    const int b   = bid >> 9;

    const int tid  = threadIdx.x;
    const int w    = tid >> 5;     // wave id (0..3): q-row strip
    const int lane = tid & 31;
    const int n    = lane & 15;    // column index within a 16x16 tile (C/B layout)
    const int lh   = lane >> 4;    // lane half

    // 1/sqrt(64) * log2(e): run the whole online softmax in base-2 so every
    // exponential is a single raw v_exp_f32.
    const float qk_scale = 0.125f * 1.44269504088896340736f;

    // ---- Q strip -> bf16 A-fragments (scale folded into Q) ----
    // A layout (16-bit, 16x32): lane m=lane&15; elem j<8 -> K=lh*8+j ; j>=8 -> K=16+lh*8+(j-8)
    const int m    = n;
    const int qrow = qi * 64 + w * 16 + m;
    const float* qg = q + (((size_t)(b * SS + qrow) * HH) + h) * DD;
    BF16Frag aq0, aq1;
    #pragma unroll
    for (int jj = 0; jj < 16; ++jj) {
        int d0 = (jj < 8) ? (lh * 8 + jj) : (16 + lh * 8 + (jj - 8));
        aq0.e[jj] = (bf16)(qg[d0]      * qk_scale);
        aq1.e[jj] = (bf16)(qg[32 + d0] * qk_scale);
    }

    // ---- running softmax state (per lane, per C-layout VGPR row r) ----
    v8f acc[4];
    float mrun[8], lrun[8];
    #pragma unroll
    for (int t = 0; t < 4; ++t) { v8f z = {0,0,0,0,0,0,0,0}; acc[t] = z; }
    #pragma unroll
    for (int r = 0; r < 8; ++r) { mrun[r] = -1e30f; lrun[r] = 0.0f; }

    const unsigned char* mrow = bmask + (((size_t)(b * HH + h) * NBQ) + qi) * NBK;

    for (int j = 0; j <= qi; ++j) {
        if (!mrow[j]) continue;                 // block-sparse skip (uniform across WG)
        __syncthreads();                        // previous tile fully consumed

        // ---- cooperative stage K,V -> LDS (fp32 -> bf16, native converts) ----
        {
            // K: thread -> (key row, 32-wide d half), contiguous global float4 loads
            const int krow = tid >> 1;
            const int dh   = (tid & 1) * 32;
            const float* kg = k + (((size_t)(b * SS + j * 64 + krow) * HH) + h) * DD + dh;
            #pragma unroll
            for (int c = 0; c < 32; c += 8) {
                float4 f0 = *(const float4*)(kg + c);
                float4 f1 = *(const float4*)(kg + c + 4);
                BF16Pack8 pk;
                pk.e[0] = (bf16)f0.x; pk.e[1] = (bf16)f0.y;
                pk.e[2] = (bf16)f0.z; pk.e[3] = (bf16)f0.w;
                pk.e[4] = (bf16)f1.x; pk.e[5] = (bf16)f1.y;
                pk.e[6] = (bf16)f1.z; pk.e[7] = (bf16)f1.w;
                *(uint4*)&kt[krow * KSTR + dh + c] = pk.q;
            }
            // V transposed: thread -> (dv row, 32-wide key half); column gather from global,
            // contiguous stores into LDS so B-fragments are contiguous b128 loads
            const int dv = tid >> 1;
            const int kh = (tid & 1) * 32;
            const float* vg = v + (((size_t)(b * SS + j * 64 + kh) * HH) + h) * DD + dv;
            #pragma unroll
            for (int kb = 0; kb < 32; kb += 8) {
                BF16Pack8 pk;
                #pragma unroll
                for (int u = 0; u < 8; ++u)
                    pk.e[u] = (bf16)vg[(size_t)(kb + u) * (HH * DD)];
                *(uint4*)&vt[dv * KSTR + kh + kb] = pk.q;
            }
        }
        __syncthreads();

        // speculative prefetch of next key block (global_prefetch_b8)
        if (j + 1 <= qi) {
            const float* nk = k + (((size_t)(b * SS + (j + 1) * 64 + (tid >> 1)) * HH) + h) * DD;
            __builtin_prefetch(nk, 0, 1);
            const float* nv = v + (((size_t)(b * SS + (j + 1) * 64 + (tid >> 1)) * HH) + h) * DD;
            __builtin_prefetch(nv, 0, 1);
        }

        const bool diag = (j == qi);

        // ---- S = Q K^T : 4 tiles x (2 chained K=32 WMMAs) ----
        v8f sc[4];
        #pragma unroll
        for (int t = 0; t < 4; ++t) {
            BF16Frag bk0, bk1;
            const int key = t * 16 + n;
            const uint4* p0 = (const uint4*)&kt[key * KSTR + lh * 8];
            bk0.q[0] = p0[0]; bk0.q[1] = p0[2];
            const uint4* p1 = (const uint4*)&kt[key * KSTR + 32 + lh * 8];
            bk1.q[0] = p1[0]; bk1.q[1] = p1[2];
            v8f z = {0,0,0,0,0,0,0,0};
            v8f c0 = __builtin_amdgcn_wmma_f32_16x16x32_bf16(false, aq0.v, false, bk0.v,
                                                             (short)0, z, false, false);
            sc[t]  = __builtin_amdgcn_wmma_f32_16x16x32_bf16(false, aq1.v, false, bk1.v,
                                                             (short)0, c0, false, false);
        }

        // causal element mask, diagonal block only (C layout: row = r + lh*8, col = n)
        if (diag) {
            #pragma unroll
            for (int t = 0; t < 4; ++t)
                #pragma unroll
                for (int r = 0; r < 8; ++r) {
                    int qloc = w * 16 + r + lh * 8;
                    int kloc = t * 16 + n;
                    if (kloc > qloc) sc[t][r] = -1e30f;
                }
        }

        // ---- online softmax, base-2 domain (row reductions across 16-lane groups) ----
        float mnew[8];
        #pragma unroll
        for (int r = 0; r < 8; ++r) {
            float mv = mrun[r];
            #pragma unroll
            for (int t = 0; t < 4; ++t) mv = fmaxf(mv, sc[t][r]);
            mv = fmaxf(mv, __shfl_xor(mv, 1, 32));
            mv = fmaxf(mv, __shfl_xor(mv, 2, 32));
            mv = fmaxf(mv, __shfl_xor(mv, 4, 32));
            mv = fmaxf(mv, __shfl_xor(mv, 8, 32));
            mnew[r] = mv;
        }
        #pragma unroll
        for (int r = 0; r < 8; ++r) {
            float a = __builtin_amdgcn_exp2f(mrun[r] - mnew[r]);   // raw v_exp_f32
            mrun[r] = mnew[r];
            float s0 = 0.0f;
            #pragma unroll
            for (int t = 0; t < 4; ++t) {
                float p = __builtin_amdgcn_exp2f(sc[t][r] - mnew[r]); // raw v_exp_f32
                sc[t][r] = p;
                s0 += p;
            }
            s0 += __shfl_xor(s0, 1, 32);
            s0 += __shfl_xor(s0, 2, 32);
            s0 += __shfl_xor(s0, 4, 32);
            s0 += __shfl_xor(s0, 8, 32);
            lrun[r] = lrun[r] * a + s0;
            #pragma unroll
            for (int t = 0; t < 4; ++t) acc[t][r] *= a;
        }

        // ---- re-layout P: C-layout f32 -> A-layout bf16 via per-wave LDS staging ----
        bf16* pw = &pst[w * 16 * PSTR];
        #pragma unroll
        for (int t = 0; t < 4; ++t)
            #pragma unroll
            for (int r = 0; r < 8; ++r)
                pw[(r + lh * 8) * PSTR + t * 16 + n] = (bf16)sc[t][r];

        BF16Frag ap0, ap1;
        const bf16* pr = pw + m * PSTR;
        ap0.q[0] = *(const uint4*)&pr[lh * 8];
        ap0.q[1] = *(const uint4*)&pr[16 + lh * 8];
        ap1.q[0] = *(const uint4*)&pr[32 + lh * 8];
        ap1.q[1] = *(const uint4*)&pr[48 + lh * 8];

        // ---- O += P V : 4 dv tiles x (2 chained K=32 WMMAs) ----
        #pragma unroll
        for (int t = 0; t < 4; ++t) {
            BF16Frag bv0, bv1;
            const int dv = t * 16 + n;
            const uint4* p0 = (const uint4*)&vt[dv * KSTR + lh * 8];
            bv0.q[0] = p0[0]; bv0.q[1] = p0[2];
            const uint4* p1 = (const uint4*)&vt[dv * KSTR + 32 + lh * 8];
            bv1.q[0] = p1[0]; bv1.q[1] = p1[2];
            acc[t] = __builtin_amdgcn_wmma_f32_16x16x32_bf16(false, ap0.v, false, bv0.v,
                                                             (short)0, acc[t], false, false);
            acc[t] = __builtin_amdgcn_wmma_f32_16x16x32_bf16(false, ap1.v, false, bv1.v,
                                                             (short)0, acc[t], false, false);
        }
    }

    // ---- epilogue: O = acc / l, written per C-layout lane mapping ----
    #pragma unroll
    for (int r = 0; r < 8; ++r) {
        const int row = qi * 64 + w * 16 + r + lh * 8;
        const float inv = 1.0f / lrun[r];
        float* og = out + (((size_t)(b * SS + row) * HH) + h) * DD;
        #pragma unroll
        for (int t = 0; t < 4; ++t)
            og[t * 16 + n] = acc[t][r] * inv;
    }
}

extern "C" void kernel_launch(void* const* d_in, const int* in_sizes, int n_in,
                              void* d_out, int out_size, void* d_ws, size_t ws_size,
                              hipStream_t stream) {
    (void)in_sizes; (void)n_in; (void)out_size; (void)d_ws; (void)ws_size;
    const float*         q    = (const float*)d_in[0];
    const float*         k    = (const float*)d_in[1];
    const float*         v    = (const float*)d_in[2];
    const unsigned char* mask = (const unsigned char*)d_in[3];
    float*               out  = (float*)d_out;

    dim3 grid(BB * HH * NBQ);   // 1024 workgroups: (b, h, q-block)
    dim3 block(128);            // 4 waves x 32
    bsattn_fa_wmma_kernel<<<grid, block, 0, stream>>>(q, k, v, mask, out);
}
